// Swarm_73212012528002
// MI455X (gfx1250) — matrix operand order repaired and checked by reading, not compiled, not run on hardware
//
#include <hip/hip_runtime.h>
#include <hip/hip_bf16.h>

#define Nn   64
#define Bv   1024
#define Sv   4
#define OBSD 128
#define HIDD 256
#define LATD 128
#define ACTD 16
#define Mv   8
#define Vv   32
#define NHv  4
#define HDv  64
#define MVv  256

typedef __attribute__((ext_vector_type(16))) __bf16 v16bf;
typedef __attribute__((ext_vector_type(8)))  __bf16 v8bf;
typedef __attribute__((ext_vector_type(8)))  float  v8f;

// ---------------- WMMA helpers (gfx1250, wave32) ----------------
__device__ __forceinline__ v8f wmma_bf16(v16bf a, v16bf b, v8f c) {
  return __builtin_amdgcn_wmma_f32_16x16x32_bf16(false, a, false, b, (short)0, c,
                                                 false, false);
}

// A fragment (16x32 bf16, MxK): lane row = lane&15, elems 0..7 -> K klo..klo+7,
// elems 8..15 -> K 16+klo.. with klo = (lane>>4)*8.
__device__ __forceinline__ v16bf load_a_frag(const __bf16* base, int rs) {
  int lane = threadIdx.x & 31;
  const __bf16* p = base + (lane & 15) * rs + ((lane >> 4) << 3);
  v8bf lo = *(const v8bf*)p;
  v8bf hi = *(const v8bf*)(p + 16);
  return __builtin_shufflevector(lo, hi, 0,1,2,3,4,5,6,7,8,9,10,11,12,13,14,15);
}

// B fragment (32x16 bf16, KxN): lane col = lane&15, elems 0..7 -> K kb..kb+7,
// elems 8..15 -> K 16+kb.. with kb = (lane>>4)*8. LDS rows [n][k] contiguous in k.
__device__ __forceinline__ v16bf load_b_frag(const __bf16* base, int rs) {
  int lane = threadIdx.x & 31;
  const __bf16* p = base + (lane & 15) * rs + ((lane >> 4) << 3);
  v8bf lo = *(const v8bf*)p;
  v8bf hi = *(const v8bf*)(p + 16);
  return __builtin_shufflevector(lo, hi, 0,1,2,3,4,5,6,7,8,9,10,11,12,13,14,15);
}

__device__ __forceinline__ v8bf pack8(float4 a, float4 b) {
  v8bf v = { (__bf16)a.x, (__bf16)a.y, (__bf16)a.z, (__bf16)a.w,
             (__bf16)b.x, (__bf16)b.y, (__bf16)b.z, (__bf16)b.w };
  return v;
}

// ---------------- generic species GEMM: Y = act(X @ W[sp]^T + b) -------------
// X: (N,B,I) f32   W: (S,O,I) f32 (wstr = O*I, 0 => shared)   Bb: (S,O) or null
// Y: (N,B,O) f32.  Block = 256 thr (8 waves); tile 64(batch) x 64(out), K = 32.
// Wave (wm,wn): wm batch sub-tile (4x16), wn 32-wide out half; 2 accs per wave.
#define BT 64
#define OTILE 64
#define KT 32
#define RS (KT + 8)   // bf16 row stride: 80B, 16B-aligned, bank-skewed
__global__ void species_gemm(const float* __restrict__ X, const float* __restrict__ W,
                             const float* __restrict__ Bb, const int* __restrict__ sp,
                             float* __restrict__ Y, int I, int O, int act,
                             int wstr, int bstr, int btiles) {
  __shared__ __align__(16) __bf16 Xs[BT][RS];
  __shared__ __align__(16) __bf16 Ws[OTILE][RS];

  int n  = blockIdx.x / btiles;
  int bt = blockIdx.x - n * btiles;
  int o0 = blockIdx.y * OTILE;
  int s  = sp ? sp[n] : 0;
  const float* Wp = W + s * wstr;
  const float* Xp = X + (n * Bv + bt * BT) * I;

  int tid  = threadIdx.x;
  int wave = tid >> 5, lane = tid & 31;
  int wm = wave & 3;     // batch sub-tile (4 x 16 = 64)
  int wn = wave >> 2;    // out half (2 x 32 = 64)

  // staging: thread -> (row, 8 consecutive cols); 64 rows x 4 threads/row
  int srow = tid >> 2;
  int scol = (tid & 3) * 8;
  int orow = min(o0 + srow, O - 1);          // clamp, no divergence
  const float* xsrc = Xp + srow * I + scol;
  const float* wsrc = Wp + orow * I + scol;

  v8f acc0 = {}, acc1 = {};
  for (int k0 = 0; k0 < I; k0 += KT) {
    float4 xa = ((const float4*)(xsrc + k0))[0];
    float4 xb = ((const float4*)(xsrc + k0))[1];
    float4 wa = ((const float4*)(wsrc + k0))[0];
    float4 wb = ((const float4*)(wsrc + k0))[1];
    if (k0 + KT < I) {
      __builtin_prefetch(xsrc + k0 + KT, 0, 1);
      __builtin_prefetch(wsrc + k0 + KT, 0, 1);
    }
    *(v8bf*)&Xs[srow][scol] = pack8(xa, xb);
    *(v8bf*)&Ws[srow][scol] = pack8(wa, wb);
    __syncthreads();
    v16bf a  = load_a_frag(&Xs[wm * 16][0], RS);
    v16bf b0 = load_b_frag(&Ws[wn * 32][0], RS);
    v16bf b1 = load_b_frag(&Ws[wn * 32 + 16][0], RS);
    acc0 = wmma_bf16(a, b0, acc0);
    acc1 = wmma_bf16(a, b1, acc1);
    __syncthreads();
  }

  int rbase = bt * BT + wm * 16 + ((lane >> 4) << 3);
  int obase = o0 + wn * 32 + (lane & 15);
#pragma unroll
  for (int half = 0; half < 2; half++) {
    int ocol = obase + half * 16;
    v8f acc = half ? acc1 : acc0;
    if (ocol < O) {
      float bias = Bb ? Bb[s * bstr + ocol] : 0.f;
#pragma unroll
      for (int r = 0; r < 8; r++) {
        float v = acc[r] + bias;
        if (act == 1) v = fmaxf(v, 0.f);
        else if (act == 2) v = tanhf(v);
        Y[(n * Bv + rbase + r) * O + ocol] = v;
      }
    }
  }
}

// ---------------- elementwise kernels ----------------
__device__ __forceinline__ float sigm(float x) { return 1.f / (1.f + __expf(-x)); }

__global__ void lstm_elem(const float* __restrict__ Gi, const float* __restrict__ Gg,
                          const float* __restrict__ Go, const float* __restrict__ bih,
                          const float* __restrict__ bhh, float* __restrict__ H) {
  int idx = blockIdx.x * 256 + threadIdx.x;          // N*B*HID threads
  int c = idx & (HIDD - 1);
  float i = sigm(Gi[idx] + bih[c] + bhh[c]);
  float g = tanhf(Gg[idx] + bih[2 * HIDD + c] + bhh[2 * HIDD + c]);
  float o = sigm(Go[idx] + bih[3 * HIDD + c] + bhh[3 * HIDD + c]);
  float cs = i * g;
  H[idx] = o * tanhf(cs);
}

__global__ void group_softmax(float* __restrict__ X) {   // (N*B*M) groups of V=32
  int idx = blockIdx.x * 256 + threadIdx.x;
  float* p = X + idx * Vv;
  float m = -3.4e38f;
  float e[Vv];
#pragma unroll
  for (int c = 0; c < Vv; c++) { e[c] = p[c]; m = fmaxf(m, e[c]); }
  float s = 0.f;
#pragma unroll
  for (int c = 0; c < Vv; c++) { e[c] = __expf(e[c] - m); s += e[c]; }
  float inv = 1.f / s;
#pragma unroll
  for (int c = 0; c < Vv; c++) p[c] = e[c] * inv;
}

// in-place LN over 256 cols; one wave per row; optional per-agent add vector.
__global__ void layernorm256(float* __restrict__ X, const float* __restrict__ g,
                             const float* __restrict__ b, const float* __restrict__ addv,
                             int rowsPerAgent) {
  int row  = blockIdx.x * 8 + (threadIdx.x >> 5);
  int lane = threadIdx.x & 31;
  float* p = X + row * HIDD;
  float vals[8];
  float s = 0.f;
#pragma unroll
  for (int t = 0; t < 8; t++) { vals[t] = p[lane + t * 32]; s += vals[t]; }
  for (int m = 16; m; m >>= 1) s += __shfl_xor(s, m, 32);
  float mu = s * (1.f / HIDD);
  float v = 0.f;
#pragma unroll
  for (int t = 0; t < 8; t++) { float d = vals[t] - mu; v += d * d; }
  for (int m = 16; m; m >>= 1) v += __shfl_xor(v, m, 32);
  float rstd = rsqrtf(v * (1.f / HIDD) + 1e-5f);
  int n = row / rowsPerAgent;
#pragma unroll
  for (int t = 0; t < 8; t++) {
    int c = lane + t * 32;
    float y = (vals[t] - mu) * rstd * g[c] + b[c];
    if (addv) y += addv[n * HIDD + c];
    p[c] = y;
  }
}

__global__ void gather_rows(const float* __restrict__ src, const int* __restrict__ sp,
                            float* __restrict__ dst) {     // dst[n,c] = src[sp[n],c]
  int idx = blockIdx.x * 256 + threadIdx.x;
  int n = idx >> 8, c = idx & 255;
  dst[idx] = src[sp[n] * HIDD + c];
}

// qh[n,o] = (qln[n,:] . wq[o,:] + bq[o]) / sqrt(hd)   (tiny: 64x256)
__global__ void q_proj(const float* __restrict__ qln, const float* __restrict__ wq,
                       const float* __restrict__ bq, float* __restrict__ qh) {
  int idx = blockIdx.x * 256 + threadIdx.x;
  int n = idx >> 8, o = idx & 255;
  const float* x = qln + n * HIDD;
  const float* w = wq + o * HIDD;
  float s = bq[o];
  for (int c = 0; c < HIDD; c++) s += x[c] * w[c];
  qh[idx] = s * 0.125f;    // 1/sqrt(64)
}

// ---------------- fused 64x64 masked attention per (b, head) ----------------
// qh: (N,256) pre-scaled; kh,vh: (N,B,256); mask: (N,N) int; out: (N,B,256)
#define ARS 72   // bf16 row stride: 144B, 16B-aligned
__global__ void attention64(const float* __restrict__ qh, const float* __restrict__ kh,
                            const float* __restrict__ vh, const int* __restrict__ mask,
                            float* __restrict__ out) {
  __shared__ __align__(16) __bf16 Qs[64][ARS];
  __shared__ __align__(16) __bf16 Ks[64][ARS];
  __shared__ __align__(16) __bf16 Vt[64][ARS];   // transposed: [d][nk]
  __shared__ __align__(16) __bf16 Ps[64][ARS];
  __shared__ float Sc[64][65];

  int b = blockIdx.x, h = blockIdx.y;
  int tid = threadIdx.x;
  int wave = tid >> 5, lane = tid & 31;

  // staging: thread -> (row, 16 consecutive cols); 64 rows x 4 threads/row
  int sr = tid >> 2;
  int sc = (tid & 2) ? 32 : 0;        // wait: need 4 slices of 16
  sc = (tid & 3) * 16;
  const float* qrow = qh + sr * 256 + h * 64 + sc;
  int kvbase = (sr * Bv + b) * 256 + h * 64 + sc;
#pragma unroll
  for (int j = 0; j < 2; j++) {
    float4 qa = ((const float4*)qrow)[2 * j];
    float4 qb = ((const float4*)qrow)[2 * j + 1];
    *(v8bf*)&Qs[sr][sc + j * 8] = pack8(qa, qb);
    float4 ka = ((const float4*)(kh + kvbase))[2 * j];
    float4 kb = ((const float4*)(kh + kvbase))[2 * j + 1];
    *(v8bf*)&Ks[sr][sc + j * 8] = pack8(ka, kb);
  }
#pragma unroll
  for (int j = 0; j < 4; j++) {
    float4 va = ((const float4*)(vh + kvbase))[j];
    Vt[sc + 4 * j + 0][sr] = (__bf16)va.x;
    Vt[sc + 4 * j + 1][sr] = (__bf16)va.y;
    Vt[sc + 4 * j + 2][sr] = (__bf16)va.z;
    Vt[sc + 4 * j + 3][sr] = (__bf16)va.w;
  }
  __syncthreads();

  // scores = Q K^T  (16 tiles of 16x16; 2 per wave)
#pragma unroll
  for (int i = 0; i < 2; i++) {
    int t = wave * 2 + i, tm = t >> 2, tn = t & 3;
    v8f acc = {};
#pragma unroll
    for (int k = 0; k < 2; k++) {
      v16bf a  = load_a_frag(&Qs[tm * 16][k * 32], ARS);
      v16bf bb = load_b_frag(&Ks[tn * 16][k * 32], ARS);
      acc = wmma_bf16(a, bb, acc);
    }
    int col = tn * 16 + (lane & 15);
    int rb  = tm * 16 + ((lane >> 4) << 3);
#pragma unroll
    for (int r = 0; r < 8; r++) {
      int row = rb + r;
      Sc[row][col] = (mask[row * Nn + col] != 0) ? acc[r] : -1e9f;
    }
  }
  __syncthreads();

  if (tid < 64) {
    float m = -3.4e38f;
    for (int c = 0; c < 64; c++) m = fmaxf(m, Sc[tid][c]);
    float s = 0.f;
    for (int c = 0; c < 64; c++) { float e = __expf(Sc[tid][c] - m); s += e; Sc[tid][c] = e; }
    float inv = 1.f / s;
    for (int c = 0; c < 64; c++) Ps[tid][c] = (__bf16)(Sc[tid][c] * inv);
  }
  __syncthreads();

  // O = P V
#pragma unroll
  for (int i = 0; i < 2; i++) {
    int t = wave * 2 + i, tm = t >> 2, tn = t & 3;
    v8f acc = {};
#pragma unroll
    for (int k = 0; k < 2; k++) {
      v16bf a  = load_a_frag(&Ps[tm * 16][k * 32], ARS);
      v16bf bb = load_b_frag(&Vt[tn * 16][k * 32], ARS);
      acc = wmma_bf16(a, bb, acc);
    }
    int d  = tn * 16 + (lane & 15);
    int rb = tm * 16 + ((lane >> 4) << 3);
#pragma unroll
    for (int r = 0; r < 8; r++) {
      int nq = rb + r;
      out[(nq * Bv + b) * 256 + h * 64 + d] = acc[r];
    }
  }
}

// ---------------- host launcher ----------------
extern "C" void kernel_launch(void* const* d_in, const int* in_sizes, int n_in,
                              void* d_out, int out_size, void* d_ws, size_t ws_size,
                              hipStream_t stream) {
  const float* obs  = (const float*)d_in[0];
  const int*   sp   = (const int*)d_in[1];
  const int*   mask = (const int*)d_in[2];
  const float* ah_w1 = (const float*)d_in[3],  * ah_b1 = (const float*)d_in[4];
  const float* ah_w2 = (const float*)d_in[5],  * ah_b2 = (const float*)d_in[6];
  const float* enc_in_w = (const float*)d_in[7], * enc_in_b = (const float*)d_in[8];
  const float* lstm_wih = (const float*)d_in[9];
  const float* lstm_bih = (const float*)d_in[11], * lstm_bhh = (const float*)d_in[12];
  const float* enc_out_w = (const float*)d_in[13], * enc_out_b = (const float*)d_in[14];
  const float* q_b = (const float*)d_in[16];
  const float* k_w = (const float*)d_in[17], * k_b = (const float*)d_in[18];
  const float* v_w = (const float*)d_in[19], * v_b = (const float*)d_in[20];
  const float* ln_q_g = (const float*)d_in[21], * ln_q_b = (const float*)d_in[22];
  const float* ln_k_g = (const float*)d_in[23], * ln_k_b = (const float*)d_in[24];
  const float* ln_v_g = (const float*)d_in[25], * ln_v_b = (const float*)d_in[26];
  const float* ln_o_g = (const float*)d_in[27], * ln_o_b = (const float*)d_in[28];
  const float* mha_in_w = (const float*)d_in[29], * mha_in_b = (const float*)d_in[30];
  const float* mha_out_w = (const float*)d_in[31], * mha_out_b = (const float*)d_in[32];
  const float* attn_out_w = (const float*)d_in[33], * attn_out_b = (const float*)d_in[34];
  const float* sender_emb = (const float*)d_in[35];
  const float* wp_w1 = (const float*)d_in[36], * wp_b1 = (const float*)d_in[37];
  const float* wp_w2 = (const float*)d_in[38], * wp_b2 = (const float*)d_in[39];
  const float* tr_w1 = (const float*)d_in[40], * tr_b1 = (const float*)d_in[41];
  const float* tr_w2 = (const float*)d_in[42], * tr_b2 = (const float*)d_in[43];

  float* out = (float*)d_out;
  const int  rows = Nn * Bv;                       // 65536
  const long big  = (long)rows * HIDD;             // floats per (N,B,256) buffer
  float* buf1 = (float*)d_ws;
  float* buf2 = buf1 + big;
  float* buf3 = buf2 + big;
  float* buf4 = buf3 + big;
  float* qtmp = buf4 + big;                        // (64,256)
  float* qh   = qtmp + Nn * HIDD;                  // (64,256)

  const int btiles = Bv / BT;                      // 16
  auto gemm = [&](const float* X, const float* W, const float* Bb, const int* spp,
                  float* Y, int I, int O, int act, int wstr, int bstr) {
    dim3 grid(Nn * btiles, (O + OTILE - 1) / OTILE);
    species_gemm<<<grid, 256, 0, stream>>>(X, W, Bb, spp, Y, I, O, act, wstr, bstr, btiles);
  };

  // 1) ActionHead
  gemm(obs,  ah_w1, ah_b1, sp, buf1, OBSD, HIDD, 1, HIDD * OBSD, HIDD);
  gemm(buf1, ah_w2, ah_b2, sp, out,  HIDD, ACTD, 2, ACTD * HIDD, ACTD);

  // 2) LSTM encoder (f-gate dead since c0 == 0)
  gemm(obs, enc_in_w, enc_in_b, sp, buf1, OBSD, HIDD, 1, HIDD * OBSD, HIDD);
  gemm(buf1, lstm_wih,                   nullptr, nullptr, buf2, HIDD, HIDD, 0, 0, 0); // i
  gemm(buf1, lstm_wih + 2 * HIDD * HIDD, nullptr, nullptr, buf3, HIDD, HIDD, 0, 0, 0); // g
  gemm(buf1, lstm_wih + 3 * HIDD * HIDD, nullptr, nullptr, buf4, HIDD, HIDD, 0, 0, 0); // o
  lstm_elem<<<(rows * HIDD) / 256, 256, 0, stream>>>(buf2, buf3, buf4, lstm_bih, lstm_bhh, buf1);

  // 3) logits -> messages (softmax over V within each of M groups)
  gemm(buf1, enc_out_w, enc_out_b, sp, buf2, HIDD, MVv, 0, MVv * HIDD, MVv);
  group_softmax<<<(rows * Mv) / 256, 256, 0, stream>>>(buf2);

  // 4) K / V paths: species proj + LN (+sender embedding on K)
  gemm(buf2, k_w, k_b, sp, buf3, MVv, HIDD, 0, HIDD * MVv, HIDD);
  layernorm256<<<rows / 8, 256, 0, stream>>>(buf3, ln_k_g, ln_k_b, sender_emb, Bv);
  gemm(buf2, v_w, v_b, sp, buf4, MVv, HIDD, 0, HIDD * MVv, HIDD);
  layernorm256<<<rows / 8, 256, 0, stream>>>(buf4, ln_v_g, ln_v_b, nullptr, Bv);

  // 5) MHA in-projections (shared weights)
  gemm(buf3, mha_in_w + 1 * HIDD * HIDD, mha_in_b + HIDD,     nullptr, buf1, HIDD, HIDD, 0, 0, 0);
  gemm(buf4, mha_in_w + 2 * HIDD * HIDD, mha_in_b + 2 * HIDD, nullptr, buf2, HIDD, HIDD, 0, 0, 0);

  // 6) Q path: batch-invariant -> per agent only
  gather_rows<<<(Nn * HIDD) / 256, 256, 0, stream>>>(q_b, sp, qtmp);
  layernorm256<<<Nn / 8, 256, 0, stream>>>(qtmp, ln_q_g, ln_q_b, nullptr, 1);
  q_proj<<<(Nn * HIDD) / 256, 256, 0, stream>>>(qtmp, mha_in_w, mha_in_b, qh);

  // 7) fused masked attention per (b, head)
  attention64<<<dim3(Bv, NHv), 256, 0, stream>>>(qh, buf1, buf2, mask, buf3);

  // 8) MHA out-proj, species out-proj + LN
  gemm(buf3, mha_out_w, mha_out_b, nullptr, buf4, HIDD, HIDD, 0, 0, 0);
  gemm(buf4, attn_out_w, attn_out_b, sp, buf1, HIDD, HIDD, 0, HIDD * HIDD, HIDD);
  layernorm256<<<rows / 8, 256, 0, stream>>>(buf1, ln_o_g, ln_o_b, nullptr, Bv);

  // 9) world-model MLPs; hat_z and hat_z_next straight into d_out
  float* hat_z      = out + (long)rows * (ACTD + LATD);  // third output
  float* hat_z_next = out + (long)rows * ACTD;           // second output
  gemm(buf1, wp_w1, wp_b1, sp, buf2, HIDD, HIDD, 1, HIDD * HIDD, HIDD);
  gemm(buf2, wp_w2, wp_b2, sp, hat_z, HIDD, LATD, 0, LATD * HIDD, LATD);
  gemm(hat_z, tr_w1, tr_b1, sp, buf1, LATD, HIDD, 1, HIDD * LATD, HIDD);
  gemm(buf1, tr_w2, tr_b2, sp, hat_z_next, HIDD, LATD, 0, LATD * HIDD, LATD);
}
